// SelfAttention_69380901699970
// MI455X (gfx1250) — matrix operand order repaired and checked by reading, not compiled, not run on hardware
//
#include <hip/hip_runtime.h>
#include <hip/hip_bf16.h>

typedef __bf16 bf16_t;
typedef __attribute__((ext_vector_type(16))) __bf16 v16bf;
typedef __attribute__((ext_vector_type(8)))  __bf16 v8bf;
typedef __attribute__((ext_vector_type(4)))  __bf16 v4bf;
typedef __attribute__((ext_vector_type(8)))  float  v8f;

#define EMBED 1024
#define SEQ   2048
#define BATCH 4
#define NHEAD 16
#define HDIM  64
#define ROWS  (BATCH * SEQ)   // 8192

// ---------------------------------------------------------------------------
// WMMA helpers (CDNA5 16x16x32 bf16, f32 accumulate)
// ---------------------------------------------------------------------------
__device__ __forceinline__ v8f wmma_bf16(v16bf a, v16bf b, v8f c) {
  // 8 args: (neg_a, A, neg_b, B, c_mod, C, reuse_a, reuse_b)
  return __builtin_amdgcn_wmma_f32_16x16x32_bf16(false, a, false, b,
                                                 (short)0, c, false, false);
}

// A-matrix fragment, 16x32 bf16, row-major source with leading dim `ld`.
// ISA layout: lanes 0-15 row M=lane, K = k0+{0..7, 16..23}; lanes 16-31 same
// rows, K = k0+{8..15, 24..31}.  => two contiguous 16B chunks per lane.
__device__ __forceinline__ v16bf load_a_frag(const bf16_t* base, int ld,
                                             int m0, int k0, int lane) {
  const int hl = lane >> 4, r = lane & 15;
  const bf16_t* p = base + (size_t)(m0 + r) * ld + (k0 + hl * 8);
  union { v16bf v; v8bf h[2]; } u;
  u.h[0] = *(const v8bf*)(p);
  u.h[1] = *(const v8bf*)(p + 16);
  return u.v;
}

// B-matrix fragment, 32x16 bf16, from a COLUMN-of-B-contiguous source:
// bt[n][k] (i.e. B transposed, row n = column n of B, contiguous in k).
// ISA layout: lanes 0-15 hold K=k0+{0..7, 16..23} of column N=lane, lanes
// 16-31 hold K=k0+{8..15, 24..31}.  => two contiguous 16B chunks per lane.
__device__ __forceinline__ v16bf load_b_frag(const bf16_t* bt, int ld,
                                             int n0, int k0, int lane) {
  const int hl = lane >> 4, r = lane & 15;
  const bf16_t* p = bt + (size_t)(n0 + r) * ld + (k0 + hl * 8);
  union { v16bf v; v8bf h[2]; } u;
  u.h[0] = *(const v8bf*)(p);
  u.h[1] = *(const v8bf*)(p + 16);
  return u.v;
}

// ---------------------------------------------------------------------------
// fp32 -> bf16 cast (vectorized)
// ---------------------------------------------------------------------------
__global__ __launch_bounds__(256) void cast_f32_bf16_kernel(
    const float* __restrict__ in, bf16_t* __restrict__ out, int n) {
  int i = (blockIdx.x * blockDim.x + threadIdx.x) * 4;
  if (i >= n) return;
  const float4 v = *(const float4*)(in + i);
  v4bf o;
  o[0] = (bf16_t)v.x; o[1] = (bf16_t)v.y; o[2] = (bf16_t)v.z; o[3] = (bf16_t)v.w;
  *(v4bf*)(out + i) = o;
}

// W [k][n] fp32  ->  Wt [n][k] bf16  (LDS tiled transpose)
__global__ __launch_bounds__(1024) void transpose_cast_kernel(
    const float* __restrict__ W, bf16_t* __restrict__ Wt) {
  __shared__ float tile[32][33];
  const int k = blockIdx.y * 32 + threadIdx.y;
  const int n = blockIdx.x * 32 + threadIdx.x;
  tile[threadIdx.y][threadIdx.x] = W[(size_t)k * EMBED + n];
  __syncthreads();
  const int nt = blockIdx.x * 32 + threadIdx.y;
  const int kt = blockIdx.y * 32 + threadIdx.x;
  Wt[(size_t)nt * EMBED + kt] = (bf16_t)tile[threadIdx.x][threadIdx.y];
}

// ---------------------------------------------------------------------------
// bf16 WMMA GEMM:  C[M,N] = A[M,K] * B[K,N] + bias
//   A: bf16 row-major [M,K];  Bt: bf16 [N,K] (B transposed)
//   mode 0: out bf16 [M,N]
//   mode 1: out bf16 Vt layout [b, n(=h*64+dh), s]  (M = b*2048+s, N = 1024)
//   mode 2: out fp32 [M,N] = C + bias + resid      (residual add)
// Workgroup = 8 waves = 128x128 tile; wave = 64x32 (4x2 WMMA tiles).
// ---------------------------------------------------------------------------
__global__ __launch_bounds__(256) void gemm_bf16_kernel(
    const bf16_t* __restrict__ A, const bf16_t* __restrict__ Bt,
    const float* __restrict__ bias, const float* __restrict__ resid,
    void* __restrict__ outp, int M, int N, int K, int mode) {
  const int lane = threadIdx.x & 31;
  const int wid  = threadIdx.x >> 5;
  const int m_w  = blockIdx.x * 128 + (wid & 1) * 64;
  const int n_w  = blockIdx.y * 128 + (wid >> 1) * 32;
  const int hl = lane >> 4, r = lane & 15;

  v8f acc[4][2];
#pragma unroll
  for (int i = 0; i < 4; ++i)
#pragma unroll
    for (int jn = 0; jn < 2; ++jn)
#pragma unroll
      for (int e = 0; e < 8; ++e) acc[i][jn][e] = 0.f;

  for (int k0 = 0; k0 < K; k0 += 32) {
    v16bf b0 = load_b_frag(Bt, K, n_w,      k0, lane);
    v16bf b1 = load_b_frag(Bt, K, n_w + 16, k0, lane);
#pragma unroll
    for (int i = 0; i < 4; ++i) {
      v16bf a = load_a_frag(A, K, m_w + i * 16, k0, lane);
      acc[i][0] = wmma_bf16(a, b0, acc[i][0]);
      acc[i][1] = wmma_bf16(a, b1, acc[i][1]);
    }
  }

  // Epilogue.  C layout: VGPR e, lanes 0-15 -> (M=e, N=lane); lanes 16-31 ->
  // (M=8+e, N=lane-16).
#pragma unroll
  for (int i = 0; i < 4; ++i) {
#pragma unroll
    for (int jn = 0; jn < 2; ++jn) {
      const int n = n_w + jn * 16 + r;
      const float bv = bias[n];
#pragma unroll
      for (int e = 0; e < 8; ++e) {
        const int m = m_w + i * 16 + hl * 8 + e;
        const float c = acc[i][jn][e] + bv;
        if (mode == 0) {
          ((bf16_t*)outp)[(size_t)m * N + n] = (bf16_t)c;
        } else if (mode == 1) {
          // Vt[b][h*64+dh][s]:  b = m>>11, s = m&2047
          const size_t idx = ((size_t)(m >> 11) * 1024 + n) * 2048 + (m & 2047);
          ((bf16_t*)outp)[idx] = (bf16_t)c;
        } else {
          const size_t idx = (size_t)m * N + n;
          ((float*)outp)[idx] = c + resid[idx];
        }
      }
    }
  }
}

// ---------------------------------------------------------------------------
// Flash attention: per (b,h), online softmax over key blocks of 64.
// 8 waves/WG; wave owns 16 queries; WG covers 128 queries.
//   Q,K: bf16 [b*S + s][1024] (head offset h*64), Vt: bf16 [b][h*64+dh][s]
//   ctx: bf16 [b*S + s][1024]
// ---------------------------------------------------------------------------
__global__ __launch_bounds__(256) void flash_attn_kernel(
    const bf16_t* __restrict__ Q, const bf16_t* __restrict__ K,
    const bf16_t* __restrict__ Vt, bf16_t* __restrict__ ctx) {
  __shared__ bf16_t p_lds[8][16][64];   // per-wave P tile (C->A relayout)

  const int lane = threadIdx.x & 31, wid = threadIdx.x >> 5;
  const int hl = lane >> 4, r = lane & 15;
  const int q0 = blockIdx.x * 128 + wid * 16;
  const int h  = blockIdx.y;
  const int b  = blockIdx.z;

  const bf16_t* Qbase  = Q  + (size_t)b * SEQ * EMBED + h * HDIM;
  const bf16_t* Kbase  = K  + (size_t)b * SEQ * EMBED + h * HDIM;
  const bf16_t* Vtbase = Vt + ((size_t)b * EMBED + h * HDIM) * SEQ;

  // Q fragments are loop-invariant: 16 queries x 64 dims = 2 k-steps
  const v16bf qa0 = load_a_frag(Qbase, EMBED, q0, 0,  lane);
  const v16bf qa1 = load_a_frag(Qbase, EMBED, q0, 32, lane);

  v8f o[4];
#pragma unroll
  for (int t = 0; t < 4; ++t)
#pragma unroll
    for (int e = 0; e < 8; ++e) o[t][e] = 0.f;
  float mrow[8], lrow[8];
#pragma unroll
  for (int e = 0; e < 8; ++e) { mrow[e] = -3.0e38f; lrow[e] = 0.f; }

  for (int kb = 0; kb < SEQ; kb += 64) {
    // S = Q * K^T  (4 key tiles of 16)
    v8f s[4];
#pragma unroll
    for (int t = 0; t < 4; ++t) {
      v8f c;
#pragma unroll
      for (int e = 0; e < 8; ++e) c[e] = 0.f;
      c = wmma_bf16(qa0, load_b_frag(Kbase, EMBED, kb + t * 16, 0,  lane), c);
      c = wmma_bf16(qa1, load_b_frag(Kbase, EMBED, kb + t * 16, 32, lane), c);
      s[t] = c;
    }

    // Online softmax.  Row (hl*8+e) is spread over the 16 lanes of this half;
    // xor-shuffles with mask<16 stay within the half.
#pragma unroll
    for (int e = 0; e < 8; ++e) {
      float mx = -3.0e38f;
#pragma unroll
      for (int t = 0; t < 4; ++t) { s[t][e] *= 0.125f; mx = fmaxf(mx, s[t][e]); }
#pragma unroll
      for (int msk = 1; msk < 16; msk <<= 1)
        mx = fmaxf(mx, __shfl_xor(mx, msk, 32));
      const float mnew = fmaxf(mrow[e], mx);
      float ps = 0.f;
#pragma unroll
      for (int t = 0; t < 4; ++t) {
        const float p = __expf(s[t][e] - mnew);
        s[t][e] = p; ps += p;
      }
#pragma unroll
      for (int msk = 1; msk < 16; msk <<= 1) ps += __shfl_xor(ps, msk, 32);
      const float rs = __expf(mrow[e] - mnew);
      lrow[e] = lrow[e] * rs + ps;
      mrow[e] = mnew;
#pragma unroll
      for (int t = 0; t < 4; ++t) o[t][e] *= rs;
    }

    // P: C-layout -> row-major LDS -> A-layout fragments
#pragma unroll
    for (int t = 0; t < 4; ++t)
#pragma unroll
      for (int e = 0; e < 8; ++e)
        p_lds[wid][hl * 8 + e][t * 16 + r] = (bf16_t)s[t][e];
    __syncthreads();   // uniform: all waves run identical trip counts

    const bf16_t* pl = &p_lds[wid][0][0];
    const v16bf pa0 = load_a_frag(pl, 64, 0, 0,  lane);
    const v16bf pa1 = load_a_frag(pl, 64, 0, 32, lane);
    // O += P * V  (4 dh tiles of 16; Vt rows are dh, contiguous in keys)
#pragma unroll
    for (int t = 0; t < 4; ++t) {
      o[t] = wmma_bf16(pa0, load_b_frag(Vtbase, SEQ, t * 16, kb,      lane), o[t]);
      o[t] = wmma_bf16(pa1, load_b_frag(Vtbase, SEQ, t * 16, kb + 32, lane), o[t]);
    }
  }

  // Normalize and store ctx (bf16)
#pragma unroll
  for (int e = 0; e < 8; ++e) {
    const float inv = 1.f / lrow[e];
#pragma unroll
    for (int t = 0; t < 4; ++t) {
      const size_t idx = (size_t)b * SEQ * EMBED +
                         (size_t)(q0 + hl * 8 + e) * EMBED + h * HDIM + t * 16 + r;
      ctx[idx] = (bf16_t)(o[t][e] * inv);
    }
  }
}

// ---------------------------------------------------------------------------
// Row LayerNorm: one 256-thread WG per row of 1024.
// ---------------------------------------------------------------------------
__global__ __launch_bounds__(256) void layernorm_kernel(
    const float* __restrict__ hbuf, const float* __restrict__ gamma,
    const float* __restrict__ beta, float* __restrict__ out) {
  const int row = blockIdx.x;
  const float* hr = hbuf + (size_t)row * EMBED;
  float vals[4], sum = 0.f, sq = 0.f;
#pragma unroll
  for (int i = 0; i < 4; ++i) {
    const float v = hr[threadIdx.x + i * 256];
    vals[i] = v; sum += v; sq += v * v;
  }
#pragma unroll
  for (int m = 1; m < 32; m <<= 1) {
    sum += __shfl_xor(sum, m, 32);
    sq  += __shfl_xor(sq,  m, 32);
  }
  __shared__ float s1[8], s2[8];
  const int lane = threadIdx.x & 31, wid = threadIdx.x >> 5;
  if (lane == 0) { s1[wid] = sum; s2[wid] = sq; }
  __syncthreads();
  sum = 0.f; sq = 0.f;
#pragma unroll
  for (int w = 0; w < 8; ++w) { sum += s1[w]; sq += s2[w]; }
  const float mu  = sum * (1.f / EMBED);
  const float var = sq * (1.f / EMBED) - mu * mu;
  const float inv = rsqrtf(var + 1e-5f);
#pragma unroll
  for (int i = 0; i < 4; ++i) {
    const int c = threadIdx.x + i * 256;
    out[(size_t)row * EMBED + c] = (vals[i] - mu) * inv * gamma[c] + beta[c];
  }
}

// ---------------------------------------------------------------------------
extern "C" void kernel_launch(void* const* d_in, const int* in_sizes, int n_in,
                              void* d_out, int out_size, void* d_ws, size_t ws_size,
                              hipStream_t stream) {
  (void)in_sizes; (void)n_in; (void)out_size; (void)ws_size;
  const float* x     = (const float*)d_in[0];
  const float* Wq    = (const float*)d_in[1];
  const float* bq    = (const float*)d_in[2];
  const float* Wk    = (const float*)d_in[3];
  const float* bk    = (const float*)d_in[4];
  const float* Wv    = (const float*)d_in[5];
  const float* bv    = (const float*)d_in[6];
  const float* Wo    = (const float*)d_in[7];
  const float* bo    = (const float*)d_in[8];
  const float* gamma = (const float*)d_in[9];
  const float* beta  = (const float*)d_in[10];
  float* out = (float*)d_out;

  char* ws = (char*)d_ws;
  size_t off = 0;
  auto alloc = [&](size_t bytes) -> void* {
    void* p = ws + off;
    off = (off + bytes + 255) & ~(size_t)255;
    return p;
  };

  bf16_t* xb   = (bf16_t*)alloc((size_t)ROWS * EMBED * 2);
  bf16_t* wqt  = (bf16_t*)alloc((size_t)EMBED * EMBED * 2);
  bf16_t* wkt  = (bf16_t*)alloc((size_t)EMBED * EMBED * 2);
  bf16_t* wvt  = (bf16_t*)alloc((size_t)EMBED * EMBED * 2);
  bf16_t* wot  = (bf16_t*)alloc((size_t)EMBED * EMBED * 2);
  bf16_t* qb   = (bf16_t*)alloc((size_t)ROWS * EMBED * 2);
  bf16_t* kb   = (bf16_t*)alloc((size_t)ROWS * EMBED * 2);
  bf16_t* vtb  = (bf16_t*)alloc((size_t)ROWS * EMBED * 2);
  bf16_t* ctx  = (bf16_t*)alloc((size_t)ROWS * EMBED * 2);
  float*  hbuf = (float*) alloc((size_t)ROWS * EMBED * 4);

  // 1) casts / weight transposes
  cast_f32_bf16_kernel<<<(ROWS * EMBED) / (256 * 4), 256, 0, stream>>>(
      x, xb, ROWS * EMBED);
  dim3 tg(32, 32), tb(32, 32);
  transpose_cast_kernel<<<tg, tb, 0, stream>>>(Wq, wqt);
  transpose_cast_kernel<<<tg, tb, 0, stream>>>(Wk, wkt);
  transpose_cast_kernel<<<tg, tb, 0, stream>>>(Wv, wvt);
  transpose_cast_kernel<<<tg, tb, 0, stream>>>(Wo, wot);

  // 2) QKV projections (V written transposed per head)
  dim3 ggrid(ROWS / 128, EMBED / 128);
  gemm_bf16_kernel<<<ggrid, 256, 0, stream>>>(xb, wqt, bq, nullptr, qb,
                                              ROWS, EMBED, EMBED, 0);
  gemm_bf16_kernel<<<ggrid, 256, 0, stream>>>(xb, wkt, bk, nullptr, kb,
                                              ROWS, EMBED, EMBED, 0);
  gemm_bf16_kernel<<<ggrid, 256, 0, stream>>>(xb, wvt, bv, nullptr, vtb,
                                              ROWS, EMBED, EMBED, 1);

  // 3) flash attention
  dim3 agrid(SEQ / 128, NHEAD, BATCH);
  flash_attn_kernel<<<agrid, 256, 0, stream>>>(qb, kb, vtb, ctx);

  // 4) output projection + bias + residual (fp32)
  gemm_bf16_kernel<<<ggrid, 256, 0, stream>>>(ctx, wot, bo, x, hbuf,
                                              ROWS, EMBED, EMBED, 2);

  // 5) LayerNorm
  layernorm_kernel<<<ROWS, 256, 0, stream>>>(hbuf, gamma, beta, out);
}
